// Decoder_43757126811853
// MI455X (gfx1250) — matrix operand order repaired and checked by reading, not compiled, not run on hardware
//
#include <hip/hip_runtime.h>
#include <hip/hip_bf16.h>
#include <math.h>

typedef __attribute__((ext_vector_type(16))) _Float16 v16h;
typedef __attribute__((ext_vector_type(8)))  float    v8f;
typedef __attribute__((ext_vector_type(4)))  float    f32x4;

#define NN    21
#define FF    6
#define HH    256
#define LL    64
#define OO    64
#define PP    25
#define BB    512
#define G4    1024
#define KCAT  288          /* 256 hidden + 6 x + 26 zero pad, 9 chunks of K=32 */
#define ROWS  16
#define NWG   (BB/ROWS)    /* 32 workgroups, no inter-WG sync needed */
#define THREADS 256

/* ---- workspace byte offsets (all 16B aligned) ---- */
#define WCAT_OFF    0u
#define WCAT_BYTES  ((size_t)NN*G4*KCAT*2)            /* 12,386,304 */
#define WHEAD_OFF   (WCAT_OFF + WCAT_BYTES)
#define WHEAD_BYTES ((size_t)NN*128*HH*2)             /*  1,376,256 */
#define BSUM_OFF    (WHEAD_OFF + WHEAD_BYTES)
#define BSUM_BYTES  ((size_t)NN*G4*4)
#define BHEAD_OFF   (BSUM_OFF + BSUM_BYTES)
#define BHEAD_BYTES ((size_t)NN*128*4)
#define CS_OFF      (BHEAD_OFF + BHEAD_BYTES)

/* ---- dynamic LDS carve (bytes) ---- */
#define SH_HYP    0          /* f16 [21][16][288] = 193,536 */
#define SH_GEFF   193536     /* f32 [21][21]      =   1,764 */
#define SH_QC     195328     /* f32 [21][16][4]   =   5,376 */
#define SH_YBUF   200704     /* f32 [2][16][64]   =   8,192 */
#define SH_LNRED  208896     /* f32 [8][16][2]    =   1,024 */
#define SH_TOTAL  209920

/* A fragment (16x32 f16): lane m=l&15; halves 0..7 at K+({0|8}), 8..15 at K+16+({0|8}) */
__device__ __forceinline__ v16h ldA(const _Float16* p) {
    union { v16h v; f32x4 f[2]; } u;
    u.f[0] = *(const f32x4*)p;
    u.f[1] = *(const f32x4*)(p + 16);
    return u.v;
}
/* B fragment (32x16 f16): lane col=l&15; halves 0..7 at K+({0|8}), 8..15 at K+16+({0|8}) */
__device__ __forceinline__ v16h ldB(const _Float16* p) {
    union { v16h v; f32x4 f[2]; } u;
    u.f[0] = *(const f32x4*)p;
    u.f[1] = *(const f32x4*)(p + 16);
    return u.v;
}

__device__ __forceinline__ float sigm(float x) { return 1.f / (1.f + expf(-x)); }

/* ================= prep kernels: f32 -> f16 weight repack ================= */
__global__ void prep_wcat(const float* __restrict__ W_hh, const float* __restrict__ W_ih,
                          _Float16* __restrict__ Wcat) {
    size_t idx = (size_t)blockIdx.x * blockDim.x + threadIdx.x;
    if (idx >= (size_t)NN * G4 * KCAT) return;
    int k = (int)(idx % KCAT);
    size_t no = idx / KCAT;                 /* n*1024 + o */
    float v = 0.f;
    if (k < HH)            v = W_hh[no * HH + k];
    else if (k < HH + FF)  v = W_ih[no * FF + (k - HH)];
    Wcat[idx] = (_Float16)v;
}

__global__ void prep_whead(const float* __restrict__ W_q, const float* __restrict__ W_Z,
                           _Float16* __restrict__ Wh) {
    size_t idx = (size_t)blockIdx.x * blockDim.x + threadIdx.x;
    if (idx >= (size_t)NN * 128 * HH) return;
    int k = (int)(idx % HH);
    size_t no = idx / HH;
    int o = (int)(no % 128);
    int n = (int)(no / 128);
    float v = (o < 64) ? W_q[((size_t)n * 64 + o) * HH + k]
                       : W_Z[((size_t)n * 64 + (o - 64)) * HH + k];
    Wh[idx] = (_Float16)v;
}

__global__ void prep_bias(const float* __restrict__ b_ih, const float* __restrict__ b_hh,
                          const float* __restrict__ b_q,  const float* __restrict__ b_Z,
                          float* __restrict__ bsum, float* __restrict__ bhead) {
    int idx = blockIdx.x * blockDim.x + threadIdx.x;
    if (idx < NN * G4) bsum[idx] = b_ih[idx] + b_hh[idx];
    if (idx < NN * 128) {
        int n = idx / 128, o = idx % 128;
        bhead[idx] = (o < 64) ? b_q[n * 64 + o] : b_Z[n * 64 + (o - 64)];
    }
}

/* ================= persistent decoder: 32 blocks x 256 threads ================= */
__global__ __launch_bounds__(THREADS, 1)
void decoder_main(const float* __restrict__ x,    const float* __restrict__ hin,
                  const float* __restrict__ zin,  const float* __restrict__ x_org,
                  const float* __restrict__ W_h0, const float* __restrict__ b_h0,
                  const float* __restrict__ W_c0, const float* __restrict__ b_c0,
                  const float* __restrict__ G,    const float* __restrict__ G_add,
                  const float* __restrict__ W_tq, const float* __restrict__ b_tq,
                  const float* __restrict__ W_tZ, const float* __restrict__ b_tZ,
                  const _Float16* __restrict__ Wcat, const _Float16* __restrict__ Whead,
                  const float* __restrict__ bsum, const float* __restrict__ bhead,
                  float* __restrict__ cs_ws, float* __restrict__ out) {
    extern __shared__ char smem[];
    _Float16* hyp  = (_Float16*)(smem + SH_HYP);   /* [n][m][288] f16: 0..255 hy, 256..261 x, rest 0 */
    float*    geff = (float*)(smem + SH_GEFF);
    float*    qcb  = (float*)(smem + SH_QC);
    float*    ybuf = (float*)(smem + SH_YBUF);
    float*    lnr  = (float*)(smem + SH_LNRED);

    const int tid = threadIdx.x;
    const int w   = tid >> 5;          /* wave 0..7: owns gate cols [32w,32w+32) */
    const int l   = tid & 31;
    const int lm  = l & 15;
    const int lh  = l >> 4;
    const int wg  = blockIdx.x;
    const int b0  = wg * ROWS;
    float* csg = cs_ws + (size_t)wg * NN * ROWS * HH;

    /* ---------- prologue ---------- */
    for (int idx = tid; idx < NN * NN; idx += THREADS) {
        int n = idx / NN;
        float s = 0.f;
        for (int m = 0; m < NN; m++) s += fabsf(G[n * NN + m]);
        geff[idx] = G[idx] / s + G_add[idx];
    }
    for (int idx = tid; idx < NN * ROWS * 4; idx += THREADS) {
        int n = idx >> 6, m = (idx >> 2) & 15, c = idx & 3;
        qcb[idx] = x_org[((size_t)(b0 + m) * NN + n) * 4 + c];
    }
    /* hs0 = rnn_h = x[:, -2] @ W_h0^T + b_h0 ; x_c0 = x[:, -1] ; zero pad */
    for (int f = tid; f < NN * ROWS * KCAT; f += THREADS) {
        int n = f / (ROWS * KCAT);
        int r = f % (ROWS * KCAT);
        int m = r / KCAT, k = r % KCAT;
        float v = 0.f;
        if (k < HH) {
            v = b_h0[n * HH + k];
            const float* xr = x + ((size_t)(b0 + m) * 10 + 8) * (NN * FF) + n * FF;
            const float* wr = W_h0 + ((size_t)n * HH + k) * FF;
            #pragma unroll
            for (int i = 0; i < FF; i++) v += xr[i] * wr[i];
        } else if (k < HH + FF) {
            v = x[((size_t)(b0 + m) * 10 + 9) * (NN * FF) + n * FF + (k - HH)];
        }
        hyp[f] = (_Float16)v;
    }
    /* cs0 = rnn_c = concat(z,h) @ W_c0^T + b_c0 ; thread-owned layout (same as LSTM phase) */
    #pragma unroll 1
    for (int n = 0; n < NN; n++) {
        #pragma unroll 1
        for (int p = 0; p < 16; p++) {
            int j = p & 7, s2 = p >> 3;
            int m = j + 8 * lh;
            int hc = w * 32 + s2 * 16 + lm;
            float v = b_c0[n * HH + hc];
            const float* wr = W_c0 + ((size_t)n * HH + hc) * (LL + FF);
            const float* zr = zin + ((size_t)(b0 + m) * NN + n) * LL;
            const float* hr = hin + ((size_t)(b0 + m) * NN + n) * FF;
            for (int i = 0; i < LL; i++) v += zr[i] * wr[i];
            #pragma unroll
            for (int i = 0; i < FF; i++) v += hr[i] * wr[LL + i];
            csg[((size_t)n * ROWS + m) * HH + hc] = v;
        }
    }
    __syncthreads();

    float* outZ = out + (size_t)BB * PP * NN * 4;

    /* ---------- time loop ---------- */
    #pragma unroll 1
    for (int t = 0; t < PP; t++) {
        /* ===== phase 1: per-node gate GEMM (K=288) + LSTM cell ===== */
        #pragma unroll 1
        for (int n = 0; n < NN; n++) {
            v8f zero = {};
            v8f acc[8];
            #pragma unroll
            for (int i = 0; i < 8; i++) acc[i] = zero;
            const _Float16* habase = hyp + n * (ROWS * KCAT) + lm * KCAT + (lh ? 8 : 0);
            const _Float16* wbase  = Wcat + (size_t)n * G4 * KCAT + (lh ? 8 : 0);
            #pragma unroll
            for (int kc = 0; kc < 9; kc++) {
                v16h A = ldA(habase + kc * 32);
                #pragma unroll
                for (int ct = 0; ct < 8; ct++) {
                    int gb = ct >> 1, s2 = ct & 1;
                    int o = gb * 256 + w * 32 + s2 * 16 + lm;
                    v16h Bf = ldB(wbase + (size_t)o * KCAT + kc * 32);
                    acc[ct] = __builtin_amdgcn_wmma_f32_16x16x32_f16(
                        false, A, false, Bf, (short)0, acc[ct], false, false);
                }
            }
            __syncthreads();   /* all waves done reading hyp[n] before hy_pre overwrite */
            #pragma unroll
            for (int s2 = 0; s2 < 2; s2++) {
                int hc = w * 32 + s2 * 16 + lm;
                float bi = bsum[n * G4 + 0 * 256 + hc];
                float bf = bsum[n * G4 + 1 * 256 + hc];
                float bg = bsum[n * G4 + 2 * 256 + hc];
                float bo = bsum[n * G4 + 3 * 256 + hc];
                #pragma unroll
                for (int j = 0; j < 8; j++) {
                    int m = j + 8 * lh;
                    float iv = acc[0 + s2][j] + bi;
                    float fv = acc[2 + s2][j] + bf;
                    float gv = acc[4 + s2][j] + bg;
                    float ov = acc[6 + s2][j] + bo;
                    size_t ci = ((size_t)n * ROWS + m) * HH + hc;
                    float c_new = sigm(fv) * csg[ci] + sigm(iv) * tanhf(gv);
                    csg[ci] = c_new;
                    float hy = sigm(ov) * tanhf(c_new);
                    hyp[n * (ROWS * KCAT) + m * KCAT + hc] = (_Float16)hy;
                }
            }
        }
        __syncthreads();

        /* ===== node mixing: hy <- G_eff @ hy_pre (fully in-thread, in place) ===== */
        #pragma unroll 1
        for (int p = 0; p < 16; p++) {
            int j = p & 7, s2 = p >> 3;
            int m = j + 8 * lh;
            int hc = w * 32 + s2 * 16 + lm;
            float v[NN];
            #pragma unroll
            for (int mm = 0; mm < NN; mm++)
                v[mm] = (float)hyp[mm * (ROWS * KCAT) + m * KCAT + hc];
            #pragma unroll
            for (int nn2 = 0; nn2 < NN; nn2++) {
                float a = 0.f;
                #pragma unroll
                for (int mm = 0; mm < NN; mm++) a += geff[nn2 * NN + mm] * v[mm];
                hyp[nn2 * (ROWS * KCAT) + m * KCAT + hc] = (_Float16)a;
            }
        }
        __syncthreads();

        /* ===== heads: [W_q | W_Z] GEMM + LayerNorm + quaternion tail ===== */
        #pragma unroll 1
        for (int n = 0; n < NN; n++) {
            v8f hacc = {};
            const _Float16* habase = hyp + n * (ROWS * KCAT) + lm * KCAT + (lh ? 8 : 0);
            int o = w * 16 + lm;  /* 0..127: cols 0..63 -> q head, 64..127 -> Z head */
            const _Float16* wb = Whead + ((size_t)n * 128 + o) * HH + (lh ? 8 : 0);
            #pragma unroll
            for (int kc = 0; kc < 8; kc++) {
                v16h A = ldA(habase + kc * 32);
                #pragma unroll
                for (int i = 0; i < 16; i++) {  /* y_t = leaky_relu(hy) on the fly */
                    float xf = (float)A[i];
                    A[i] = (_Float16)(xf > 0.f ? xf : 0.01f * xf);
                }
                v16h Bf = ldB(wb + kc * 32);
                hacc = __builtin_amdgcn_wmma_f32_16x16x32_f16(
                    false, A, false, Bf, (short)0, hacc, false, false);
            }
            float bb = bhead[n * 128 + o];
            float hv[8];
            #pragma unroll
            for (int j = 0; j < 8; j++) {
                float vv = hacc[j] + bb;
                hv[j] = vv;
                float s1 = vv, sq = vv * vv;
                s1 += __shfl_xor(s1, 1);  s1 += __shfl_xor(s1, 2);
                s1 += __shfl_xor(s1, 4);  s1 += __shfl_xor(s1, 8);
                sq += __shfl_xor(sq, 1);  sq += __shfl_xor(sq, 2);
                sq += __shfl_xor(sq, 4);  sq += __shfl_xor(sq, 8);
                if (lm == 0) {
                    int row = j + 8 * lh;
                    lnr[(w * 16 + row) * 2 + 0] = s1;
                    lnr[(w * 16 + row) * 2 + 1] = sq;
                }
            }
            __syncthreads();
            int head = w >> 2;
            int wb0 = w & ~3;
            #pragma unroll
            for (int j = 0; j < 8; j++) {
                int row = j + 8 * lh;
                float S = 0.f, Q = 0.f;
                #pragma unroll
                for (int ww = 0; ww < 4; ww++) {
                    S += lnr[((wb0 + ww) * 16 + row) * 2 + 0];
                    Q += lnr[((wb0 + ww) * 16 + row) * 2 + 1];
                }
                float mean = S * (1.f / 64.f);
                float var  = Q * (1.f / 64.f) - mean * mean;
                float inv  = rsqrtf(var + 1e-5f);
                float vv = (hv[j] - mean) * inv;
                vv = (head == 0) ? (vv > 0.f ? vv : 0.01f * vv) : tanhf(vv);
                ybuf[(head * 16 + row) * OO + (w & 3) * 16 + lm] = vv;
            }
            __syncthreads();
            if (tid < 16) {
                int m = tid;
                const float* yq = ybuf + m * OO;
                const float* yZ = ybuf + (16 + m) * OO;
                float dq[4]; dq[0] = 1.f;
                float Zt[3];
                #pragma unroll
                for (int c = 0; c < 3; c++) {
                    float a1 = b_tq[n * 3 + c];
                    float a2 = b_tZ[n * 3 + c];
                    const float* wq = W_tq + ((size_t)n * 3 + c) * OO;
                    const float* wz = W_tZ + ((size_t)n * 3 + c) * OO;
                    for (int o2 = 0; o2 < OO; o2++) {
                        a1 += yq[o2] * wq[o2];
                        a2 += yZ[o2] * wz[o2];
                    }
                    dq[1 + c] = a1; Zt[c] = a2;
                }
                float nrm = rsqrtf(1.f + dq[1]*dq[1] + dq[2]*dq[2] + dq[3]*dq[3]);
                #pragma unroll
                for (int c = 0; c < 4; c++) dq[c] *= nrm;
                float* qc = qcb + (n * ROWS + m) * 4;
                float bw = qc[0], bx = qc[1], by = qc[2], bz = qc[3];
                float aw = dq[0], ax = dq[1], ay = dq[2], az = dq[3];
                float q0 = aw*bw - ax*bx - ay*by - az*bz;
                float q1 = aw*bx + ax*bw + ay*bz - az*by;
                float q2 = aw*by - ax*bz + ay*bw + az*bx;
                float q3 = aw*bz + ax*by - ay*bx + az*bw;
                qc[0] = q0; qc[1] = q1; qc[2] = q2; qc[3] = q3;
                float iq  = 1.f / (1.1f - q0);
                float idq = 1.f / (1.1f - dq[0]);
                _Float16* xc = hyp + n * (ROWS * KCAT) + m * KCAT + HH; /* x_c for next step */
                xc[0] = (_Float16)(q1 * iq);
                xc[1] = (_Float16)(q2 * iq);
                xc[2] = (_Float16)(q3 * iq);
                xc[3] = (_Float16)(dq[1] * idq);
                xc[4] = (_Float16)(dq[2] * idq);
                xc[5] = (_Float16)(dq[3] * idq);
                size_t ob = (((size_t)(b0 + m) * PP + t) * NN + n);
                out[ob * 4 + 0] = q0; out[ob * 4 + 1] = q1;
                out[ob * 4 + 2] = q2; out[ob * 4 + 3] = q3;
                outZ[ob * 3 + 0] = Zt[0]; outZ[ob * 3 + 1] = Zt[1]; outZ[ob * 3 + 2] = Zt[2];
            }
            __syncthreads();  /* protect ybuf/lnr reuse by next node */
        }
    }
}

/* ================= host launcher ================= */
extern "C" void kernel_launch(void* const* d_in, const int* in_sizes, int n_in,
                              void* d_out, int out_size, void* d_ws, size_t ws_size,
                              hipStream_t stream) {
    const float* x     = (const float*)d_in[0];
    const float* h     = (const float*)d_in[1];
    const float* z     = (const float*)d_in[2];
    const float* x_org = (const float*)d_in[3];
    /* d_in[4] = y (zeros, unused) */
    const float* W_h0  = (const float*)d_in[5];
    const float* b_h0  = (const float*)d_in[6];
    const float* W_c0  = (const float*)d_in[7];
    const float* b_c0  = (const float*)d_in[8];
    const float* W_ih  = (const float*)d_in[9];
    const float* b_ih  = (const float*)d_in[10];
    const float* W_hh  = (const float*)d_in[11];
    const float* b_hh  = (const float*)d_in[12];
    const float* G     = (const float*)d_in[13];
    const float* G_add = (const float*)d_in[14];
    const float* W_q   = (const float*)d_in[15];
    const float* b_q   = (const float*)d_in[16];
    const float* W_Z   = (const float*)d_in[17];
    const float* b_Z   = (const float*)d_in[18];
    const float* W_tq  = (const float*)d_in[19];
    const float* b_tq  = (const float*)d_in[20];
    const float* W_tZ  = (const float*)d_in[21];
    const float* b_tZ  = (const float*)d_in[22];

    char* ws = (char*)d_ws;
    _Float16* Wcat  = (_Float16*)(ws + WCAT_OFF);
    _Float16* Whead = (_Float16*)(ws + WHEAD_OFF);
    float*    bsum  = (float*)(ws + BSUM_OFF);
    float*    bhead = (float*)(ws + BHEAD_OFF);
    float*    cs    = (float*)(ws + CS_OFF);

    prep_wcat<<<(int)(((size_t)NN * G4 * KCAT + 255) / 256), 256, 0, stream>>>(W_hh, W_ih, Wcat);
    prep_whead<<<(int)(((size_t)NN * 128 * HH + 255) / 256), 256, 0, stream>>>(W_q, W_Z, Whead);
    prep_bias<<<(NN * G4 + 255) / 256, 256, 0, stream>>>(b_ih, b_hh, b_q, b_Z, bsum, bhead);

    (void)hipFuncSetAttribute((const void*)decoder_main,
                              hipFuncAttributeMaxDynamicSharedMemorySize, SH_TOTAL);
    decoder_main<<<NWG, THREADS, SH_TOTAL, stream>>>(
        x, h, z, x_org, W_h0, b_h0, W_c0, b_c0, G, G_add,
        W_tq, b_tq, W_tZ, b_tZ, Wcat, Whead, bsum, bhead, cs, (float*)d_out);
}
